// QuantLinear_26087631356146
// MI455X (gfx1250) — matrix operand order repaired and checked by reading, not compiled, not run on hardware
//
#include <hip/hip_runtime.h>

typedef __attribute__((ext_vector_type(16))) _Float16 v16h;
typedef __attribute__((ext_vector_type(8)))  _Float16 v8h;
typedef __attribute__((ext_vector_type(2)))  _Float16 v2h;
typedef __attribute__((ext_vector_type(8)))  float    v8f;
typedef __attribute__((ext_vector_type(4)))  float    v4f;

#define IN_F   4096
#define OUT_F  11008
#define MROWS  64
#define KCHUNK 128                 // == quant group size
#define NGRP   (IN_F / KCHUNK)     // 32 groups
#define HGRP   (NGRP / 2)          // 16 groups per K-half
#define LDS_STRIDE 136             // 128 + 8 halves pad -> conflict-free b128
#define XBUF_H (MROWS * LDS_STRIDE)       // halves per x buffer (8704)

// Block: 256 threads = 8 waves on one WGP.
//   wave w (0..3): columns [blockIdx.x*64 + 16w, +16), K groups 0..15
//   wave w+4     : same columns,                       K groups 16..31
// Partials combined via LDS (deterministic, no float atomics).
//
// K relabeling: within every aligned 8-group of K, order P=[0,4,1,5,2,6,3,7].
// B gets this for free from the 0x000F000F nibble-pair extraction; A gets it
// by storing x into LDS pre-permuted. Dequant is exact:
//   t = (nib|0x6400) = 1024+nib ;  (t - (1024+z)) * sc  == RN(sc*(nib-z))
__global__ __launch_bounds__(256) void qlin4_wmma_kernel(
    const float* __restrict__ x,        // [64, 4096]
    const int*   __restrict__ qweight,  // [512, 11008] int4x8 along K
    const int*   __restrict__ qzeros,   // [32, 1376]   int4x8 along N
    const float* __restrict__ scales,   // [32, 11008]
    const float* __restrict__ bias,     // [11008]
    float*       __restrict__ out)      // [64, 11008]
{
    __shared__ __align__(16) unsigned char smem[2 * XBUF_H * 2]; // 34816 B
    _Float16* xs  = (_Float16*)smem;    // two chunk buffers of x (f16)
    float*    red = (float*)smem;       // reused for partial reduction (16 KB)

    const int tid  = threadIdx.x;
    const int wave = tid >> 5;
    const int lane = tid & 31;
    const int l16  = lane & 15;
    const int hi   = lane >> 4;
    const int cw   = wave & 3;                      // column sub-tile
    const int grp  = wave >> 2;                     // K-half (0 or 1)
    const int col  = blockIdx.x * 64 + cw * 16 + l16;

    v8f acc[4];
#pragma unroll
    for (int i = 0; i < 4; ++i)
#pragma unroll
        for (int j = 0; j < 8; ++j) acc[i][j] = 0.0f;

    for (int t = 0; t < HGRP; ++t) {
        __syncthreads();   // previous iteration's A reads done
        // Stage x chunks for BOTH K-halves: groups t and t+16.
        // 8 consecutive f32 per thread -> P-permuted v8h -> one b128 store.
        for (int i = tid * 8; i < 2 * MROWS * KCHUNK; i += 256 * 8) {
            const int buf = i >> 13;                // 0: group t, 1: group t+16
            const int j   = i & 8191;
            const int r   = j >> 7;
            const int c   = j & 127;                // multiple of 8
            const float* src = x + r * IN_F + (t + HGRP * buf) * KCHUNK + c;
            v4f v0 = *(const v4f*)(src);
            v4f v1 = *(const v4f*)(src + 4);
            v8h h;                                  // invP = [0,2,4,6,1,3,5,7]
            h[0] = (_Float16)v0[0]; h[2] = (_Float16)v0[1];
            h[4] = (_Float16)v0[2]; h[6] = (_Float16)v0[3];
            h[1] = (_Float16)v1[0]; h[3] = (_Float16)v1[1];
            h[5] = (_Float16)v1[2]; h[7] = (_Float16)v1[3];
            *(v8h*)(&xs[buf * XBUF_H + r * LDS_STRIDE + c]) = h;
        }
        __syncthreads();

        const int cc = t + HGRP * grp;              // this wave's quant group
        const float sc = scales[cc * OUT_F + col];
        const int   qz = qzeros[cc * (OUT_F / 8) + (col >> 3)];
        const float zf = (float)(((qz >> ((col & 7) * 4)) & 15) + 1);
        const _Float16 sch = (_Float16)sc;
        const _Float16 mzh = (_Float16)(1024.0f + zf);  // exact in f16
        const v2h sc2 = {sch, sch};
        const v2h mz2 = {mzh, mzh};

        const int rowBase = cc * 16;                // qweight row of group
        const _Float16* xb = xs + grp * XBUF_H;

#pragma unroll
        for (int s = 0; s < 4; ++s) {
            // B (32x16 f16): lane n=l16; slot k=16*hi+e holds W[pi(k)].
            const int r0 = rowBase + 4 * s + 2 * hi;
            const unsigned int w0 = (unsigned int)qweight[(r0    ) * OUT_F + col];
            const unsigned int w1 = (unsigned int)qweight[(r0 + 1) * OUT_F + col];
            if (t + 1 < HGRP) {   // next group, same sub-step
                __builtin_prefetch(&qweight[(r0 + 16) * OUT_F + col], 0, 0);
            }
            union { v16h v; v2h p[8]; } b;
#pragma unroll
            for (int i = 0; i < 4; ++i) {
                const unsigned int t0 = ((w0 >> (4 * i)) & 0x000F000Fu) | 0x64006400u;
                const unsigned int t1 = ((w1 >> (4 * i)) & 0x000F000Fu) | 0x64006400u;
                b.p[i]     = (__builtin_bit_cast(v2h, t0) - mz2) * sc2;
                b.p[i + 4] = (__builtin_bit_cast(v2h, t1) - mz2) * sc2;
            }

            // A (16x32 f16): lane m=l16; two 8-aligned b128 runs (P-permuted
            // in LDS to match B's nibble pairing).
            const int kb = 32 * s + 16 * hi;
#pragma unroll
            for (int rt = 0; rt < 4; ++rt) {
                const int m = rt * 16 + l16;
                v8h a0 = *(const v8h*)(&xb[m * LDS_STRIDE + kb]);
                v8h a1 = *(const v8h*)(&xb[m * LDS_STRIDE + kb + 8]);
                v16h a = __builtin_shufflevector(a0, a1,
                          0,1,2,3,4,5,6,7,8,9,10,11,12,13,14,15);
                acc[rt] = __builtin_amdgcn_wmma_f32_16x16x32_f16(
                    /*neg_a=*/false, a, /*neg_b=*/false, b.v,
                    /*c_mod=*/(short)0, acc[rt],
                    /*reuse_a=*/false, /*reuse_b=*/false);
            }
        }
    }

    // Combine K-halves via LDS (smem reused after barrier), add bias, store.
    __syncthreads();
    if (grp == 1) {
#pragma unroll
        for (int rt = 0; rt < 4; ++rt) {
            *(v8f*)(&red[(cw * 4 + rt) * 256 + lane * 8]) = acc[rt];
        }
    }
    __syncthreads();
    if (grp == 0) {
        const float bv = bias[col];
#pragma unroll
        for (int rt = 0; rt < 4; ++rt) {
            v8f other = *(const v8f*)(&red[(cw * 4 + rt) * 256 + lane * 8]);
#pragma unroll
            for (int v = 0; v < 8; ++v) {
                // C/D layout: VGPR v, lane: N=l16, M = v + 8*hi (+16*rt)
                const int m = rt * 16 + v + 8 * hi;
                out[m * OUT_F + col] = acc[rt][v] + other[v] + bv;
            }
        }
    }
}

extern "C" void kernel_launch(void* const* d_in, const int* in_sizes, int n_in,
                              void* d_out, int out_size, void* d_ws, size_t ws_size,
                              hipStream_t stream) {
    const float* x       = (const float*)d_in[0];
    const int*   qweight = (const int*)  d_in[1];
    const int*   qzeros  = (const int*)  d_in[2];
    const float* scales  = (const float*)d_in[3];
    // d_in[4] = g_idx: groups are contiguous (k/128), computed implicitly.
    const float* bias    = (const float*)d_in[5];
    float*       out     = (float*)d_out;

    dim3 grid(OUT_F / 64);   // 172 blocks, 64 output cols each
    dim3 block(256);         // 8 waves: 2-way K split x 4 column sub-tiles
    qlin4_wmma_kernel<<<grid, block, 0, stream>>>(x, qweight, qzeros, scales, bias, out);
}
